// QuantizedLinear_10857677324721
// MI455X (gfx1250) — compile-verified, matrix operand order checked by reading, not run on hardware
//
#include <hip/hip_runtime.h>

typedef __attribute__((ext_vector_type(16))) _Float16 v16h;
typedef __attribute__((ext_vector_type(8)))  _Float16 v8h;
typedef __attribute__((ext_vector_type(8)))  float    v8f;
typedef __attribute__((ext_vector_type(4)))  int      v4i;
typedef __attribute__((ext_vector_type(4)))  unsigned u32x4;
typedef __attribute__((ext_vector_type(8)))  int      i32x8;
typedef __attribute__((ext_vector_type(4)))  int      i32x4;

#if __has_builtin(__builtin_amdgcn_tensor_load_to_lds)
#define HAVE_TDM 1
#else
#define HAVE_TDM 0
#endif
#if __has_builtin(__builtin_amdgcn_global_load_async_to_lds_b128)
#define HAVE_ASYNC_LDS 1
#else
#define HAVE_ASYNC_LDS 0
#endif

constexpr int IN_F  = 4096;
constexpr int OUT_F = 11008;
constexpr int BM = 128, BN = 128, BK = 64;
constexpr int LSTR = 72;   // halves; 64 + 8 pad -> conflict-free b128 fragment loads

// One 16x32-per-k-step f16 WMMA operand fragment from LDS.
__device__ inline v16h load_frag(const _Float16* rowbase, int h) {
    v8h lo = *(const v8h*)(rowbase + h * 8);
    v8h hi = *(const v8h*)(rowbase + 16 + h * 8);
    return __builtin_shufflevector(lo, hi, 0, 1, 2, 3, 4, 5, 6, 7,
                                           8, 9, 10, 11, 12, 13, 14, 15);
}

// Magic-number int8 -> fp16 dequant: (b ^ 0x80) spliced into 0x64:byte fp16
// (= 1024 + b + 128), then packed add of -1152.
__device__ inline v8h cvt8(unsigned w0, unsigned w1) {
    w0 ^= 0x80808080u;
    w1 ^= 0x80808080u;
    union { unsigned u[4]; v8h v; } t;
    t.u[0] = __builtin_amdgcn_perm(0x64646464u, w0, 0x04010400u);
    t.u[1] = __builtin_amdgcn_perm(0x64646464u, w0, 0x04030402u);
    t.u[2] = __builtin_amdgcn_perm(0x64646464u, w1, 0x04010400u);
    t.u[3] = __builtin_amdgcn_perm(0x64646464u, w1, 0x04030402u);
    const _Float16 c = (_Float16)(-1152.0f);
    const v8h bias = {c, c, c, c, c, c, c, c};
    return t.v + bias;
}

__device__ inline void wait_staging() {
#if HAVE_TDM
#if __has_builtin(__builtin_amdgcn_s_wait_tensorcnt)
    __builtin_amdgcn_s_wait_tensorcnt(0);
#else
    asm volatile("s_wait_tensorcnt 0x0" ::: "memory");
#endif
#elif HAVE_ASYNC_LDS
#if __has_builtin(__builtin_amdgcn_s_wait_asynccnt)
    __builtin_amdgcn_s_wait_asynccnt(0);
#else
    asm volatile("s_wait_asynccnt 0" ::: "memory");
#endif
#endif
}

__global__ __launch_bounds__(128, 2)
void qlinear_wmma_f16(const _Float16* __restrict__ X,
                      const signed char* __restrict__ W,
                      const float* __restrict__ S,
                      _Float16* __restrict__ Y) {
    __shared__ __align__(16) _Float16 Al[2][BM * LSTR];
    __shared__ __align__(16) _Float16 Bl[2][BN * LSTR];

    const int tid   = threadIdx.x;
    const int lane  = tid & 31;
    const int wave  = tid >> 5;    // 0..3
    const int waveM = wave >> 1;   // 0..1 -> 64-row slab
    const int waveN = wave & 1;    // 0..1 -> 64-col slab
    const int r     = lane & 15;
    const int h     = lane >> 4;

    const int blockN = blockIdx.x; // 0..85
    const int blockM = blockIdx.y; // 0..63

    const _Float16*    xg = X + (size_t)(blockM * BM + tid) * IN_F;   // fallback A path
    const signed char* wg = W + (size_t)(blockN * BN + tid) * IN_F;
    _Float16* aL[2] = { &Al[0][tid * LSTR], &Al[1][tid * LSTR] };
    _Float16* bL[2] = { &Bl[0][tid * LSTR], &Bl[1][tid * LSTR] };

    v8f acc[4][4] = {};

    int4 wr[4];
#if !HAVE_TDM && !HAVE_ASYNC_LDS
    float4 ar[8];
#endif

    // Stage the 128x64-half x tile for k-slab kt into Al[sbuf].
    auto issueA = [&](int sbuf, int kt) {
#if HAVE_TDM
        // One TDM descriptor per workgroup; wave 0 issues (EXEC ignored by TDM).
        if (wave == 0) {
            const unsigned lds = (unsigned)(size_t)(&Al[sbuf][0]);
            const unsigned long long ga =
                (unsigned long long)(size_t)(X + (size_t)(blockM * BM) * IN_F + kt * BK);
            u32x4 g0;
            g0[0] = 1u;                                   // count=1, user descriptor
            g0[1] = lds;                                  // lds_addr (bytes)
            g0[2] = (unsigned)(ga & 0xFFFFFFFFu);         // global_addr[31:0]
            g0[3] = (unsigned)((ga >> 32) & 0x01FFFFFFu)  // global_addr[56:32]
                  | (2u << 30);                           // type = 2 ("image")
            i32x8 g1;
            g1[0] = (1 << 16)    // data_size = 2 bytes
                  | (1 << 20)    // pad_enable
                  | (4 << 22)    // pad_interval: every 32 DWORDs (= one 64-half row)
                  | (3 << 25);   // pad_amount: 4 DWORDs (= 8 halves -> 72-half stride)
            g1[1] = (BK & 0xFFFF) << 16;                  // tensor_dim0 = 64 (lo16)
            g1[2] = (BM & 0xFFFF) << 16;                  // dim0 hi=0 | tensor_dim1 = 128
            g1[3] = (BK & 0xFFFF) << 16;                  // dim1 hi=0 | tile_dim0 = 64
            g1[4] = BM;                                   // tile_dim1 = 128, tile_dim2 = 0
            g1[5] = IN_F;                                 // tensor_dim0_stride = 4096 (lo32)
            g1[6] = 0;                                    // stride0 hi | stride1 lo
            g1[7] = 0;                                    // stride1 hi
            const i32x4 z4 = {0, 0, 0, 0};
            const i32x8 z8 = {0, 0, 0, 0, 0, 0, 0, 0};
            __builtin_amdgcn_tensor_load_to_lds(g0, g1, z4, z4, z8, 0);
        }
#elif HAVE_ASYNC_LDS
        const _Float16* g = xg + kt * BK;
        _Float16*       l = aL[sbuf];
#pragma unroll
        for (int i = 0; i < 8; ++i)
            __builtin_amdgcn_global_load_async_to_lds_b128(
                (__attribute__((address_space(1))) v4i*)(g + i * 8),
                (__attribute__((address_space(3))) v4i*)(l + i * 8),
                0, 0);
#else
        const float4* g = (const float4*)(xg + kt * BK);
#pragma unroll
        for (int i = 0; i < 8; ++i) ar[i] = g[i];
#endif
    };
    auto storeA = [&](int sbuf) {
#if !HAVE_TDM && !HAVE_ASYNC_LDS
        float4* l = (float4*)aL[sbuf];
#pragma unroll
        for (int i = 0; i < 8; ++i) l[i] = ar[i];
#else
        (void)sbuf;
#endif
    };
    auto loadW = [&](int kt) {
        const int4* g = (const int4*)(wg + kt * BK);
#pragma unroll
        for (int i = 0; i < 4; ++i) wr[i] = g[i];
    };
    auto storeW = [&](int sbuf) {
        v8h* l = (v8h*)bL[sbuf];
#pragma unroll
        for (int i = 0; i < 4; ++i) {
            l[2 * i]     = cvt8(wr[i].x, wr[i].y);
            l[2 * i + 1] = cvt8(wr[i].z, wr[i].w);
        }
    };
    auto compute = [&](int sbuf) {
#pragma unroll
        for (int kk = 0; kk < 2; ++kk) {
            v16h af[4], bf[4];
#pragma unroll
            for (int mt = 0; mt < 4; ++mt)
                af[mt] = load_frag(&Al[sbuf][(waveM * 64 + mt * 16 + r) * LSTR + kk * 32], h);
#pragma unroll
            for (int nt = 0; nt < 4; ++nt)
                bf[nt] = load_frag(&Bl[sbuf][(waveN * 64 + nt * 16 + r) * LSTR + kk * 32], h);
#pragma unroll
            for (int mt = 0; mt < 4; ++mt)
#pragma unroll
                for (int nt = 0; nt < 4; ++nt)
                    acc[mt][nt] = __builtin_amdgcn_wmma_f32_16x16x32_f16(
                        false, af[mt], false, bf[nt],
                        (short)0, acc[mt][nt], false, false);
        }
    };

    // Prologue: stage k-slab 0 into buffer 0.
    issueA(0, 0);
    loadW(0);
    storeW(0);
    storeA(0);
    wait_staging();            // tensorcnt/asynccnt==0 is a no-op for non-issuing waves
    __syncthreads();

    constexpr int NK = IN_F / BK;  // 64
    int buf = 0;
    for (int kt = 0; kt < NK; ++kt) {
        if (kt + 1 < NK) {
            issueA(buf ^ 1, kt + 1);   // buf^1 free: barrier at end of kt-1 passed
            loadW(kt + 1);
        }
        compute(buf);                  // 32 WMMAs
        if (kt + 1 < NK) {
            storeW(buf ^ 1);
            storeA(buf ^ 1);
            wait_staging();
            __syncthreads();
            buf ^= 1;
        }
    }

    // Epilogue: per-output-channel fp32 scale, fp16 store.
#pragma unroll
    for (int nt = 0; nt < 4; ++nt) {
        const int n = blockN * BN + waveN * 64 + nt * 16 + r;
        const float s = S[n];
#pragma unroll
        for (int mt = 0; mt < 4; ++mt) {
#pragma unroll
            for (int v = 0; v < 8; ++v) {
                const int m = blockM * BM + waveM * 64 + mt * 16 + h * 8 + v;
                Y[(size_t)m * OUT_F + n] = (_Float16)(acc[mt][nt][v] * s);
            }
        }
    }
}

extern "C" void kernel_launch(void* const* d_in, const int* in_sizes, int n_in,
                              void* d_out, int out_size, void* d_ws, size_t ws_size,
                              hipStream_t stream) {
    const _Float16*    X = (const _Float16*)d_in[0];
    const signed char* W = (const signed char*)d_in[1];
    const float*       S = (const float*)d_in[2];
    _Float16*          Y = (_Float16*)d_out;

    const int T = in_sizes[0] / IN_F;          // 8192
    dim3 grid(OUT_F / BN, T / BM);             // 86 x 64
    qlinear_wmma_f16<<<grid, 128, 0, stream>>>(X, W, S, Y);
}
